// CompoundProteinInteractionPrediction_4733053960501
// MI455X (gfx1250) — compile-verified, matrix operand order checked by reading, not run on hardware
//
#include <hip/hip_runtime.h>

typedef __attribute__((ext_vector_type(16))) _Float16 v16h;
typedef __attribute__((ext_vector_type(8)))  float    v8f;

#define DIM 128
#define NN  4096
#define MM  4096

// ---------------------------------------------------------------------------
// WMMA fragment loader for C = A * B^T with A,B row-major f16.
// 16x32 f16 A-fragment layout (ISA 7.12.2): lanes 0-15 hold rows 0-15 with
// K = [0..7] in VGPR0-3 and K = [16..23] in VGPR4-7; lanes 16-31 hold the same
// rows with K = [8..15] and [24..31]. B (32x16) is the symmetric layout with
// lane = output column, so the same loader serves both operands.
// ---------------------------------------------------------------------------
__device__ __forceinline__ v16h frag_load(const _Float16* __restrict__ tile, int ld) {
    const int lane = threadIdx.x & 31;
    const int r    = lane & 15;
    const int kb   = (lane & 16) ? 8 : 0;
    const _Float16* q = tile + (size_t)r * (size_t)ld + kb;
    v16h f;
    uint4* fu = reinterpret_cast<uint4*>(&f);
    fu[0] = *reinterpret_cast<const uint4*>(q);       // K = kb .. kb+7
    fu[1] = *reinterpret_cast<const uint4*>(q + 16);  // K = kb+16 .. kb+23
    return f;
}

// C/D fragment mapping: VGPR r, lanes 0-15 -> (row=r, col=lane);
//                       lanes 16-31 -> (row=r+8, col=lane-16).
__device__ __forceinline__ void frag_rc(int& rbase, int& col) {
    const int lane = threadIdx.x & 31;
    col   = lane & 15;
    rbase = (lane & 16) ? 8 : 0;
}

// ---------------------------------------------------------------------------
// Generic GEMM: C[M,N] = act(A[M,K] * B[N,K]^T + bias[N]); optional f32 and
// f16 outputs. One wave (32 threads) per 16x16 output tile.
// ---------------------------------------------------------------------------
__global__ void gemm_bias_act(const _Float16* __restrict__ A, const _Float16* __restrict__ B,
                              const float* __restrict__ bias,
                              float* __restrict__ Cf, _Float16* __restrict__ Ch,
                              int K, int lda, int ldb, int ldc, int do_relu) {
    const int i0 = blockIdx.x * 16;
    const int j0 = blockIdx.y * 16;
    v8f acc = {};
    for (int k = 0; k < K; k += 32) {
        v16h a = frag_load(A + (size_t)i0 * lda + k, lda);
        v16h b = frag_load(B + (size_t)j0 * ldb + k, ldb);
        acc = __builtin_amdgcn_wmma_f32_16x16x32_f16(false, a, false, b, (short)0, acc,
                                                     false, false);
    }
    int rb, col; frag_rc(rb, col);
    const float bv = bias ? bias[j0 + col] : 0.f;
#pragma unroll
    for (int r = 0; r < 8; ++r) {
        const int row = i0 + rb + r;
        float v = acc[r] + bv;
        if (do_relu) v = fmaxf(v, 0.f);
        const size_t idx = (size_t)row * ldc + j0 + col;
        if (Cf) Cf[idx] = v;
        if (Ch) Ch[idx] = (_Float16)v;
    }
}

// ---------------------------------------------------------------------------
// GAT: attn = (hs * hs^T) .* adjacency, written as f16.
// ---------------------------------------------------------------------------
__global__ void gemm_attn(const _Float16* __restrict__ hs, const float* __restrict__ adj,
                          _Float16* __restrict__ attn) {
    const int i0 = blockIdx.x * 16;
    const int j0 = blockIdx.y * 16;
    v8f acc = {};
#pragma unroll
    for (int k = 0; k < DIM; k += 32) {
        v16h a = frag_load(hs + (size_t)i0 * DIM + k, DIM);
        v16h b = frag_load(hs + (size_t)j0 * DIM + k, DIM);
        acc = __builtin_amdgcn_wmma_f32_16x16x32_f16(false, a, false, b, (short)0, acc,
                                                     false, false);
    }
    int rb, col; frag_rc(rb, col);
#pragma unroll
    for (int r = 0; r < 8; ++r) {
        const int row = i0 + rb + r;
        const size_t idx = (size_t)row * NN + j0 + col;
        attn[idx] = (_Float16)(acc[r] * adj[idx]);
    }
}

// denom[i] = max(sum_j |attn[i][j]|, eps)
__global__ void row_abs_sum(const _Float16* __restrict__ attn, float* __restrict__ denom) {
    __shared__ float red[256];
    const int i = blockIdx.x;
    float s = 0.f;
    for (int j = threadIdx.x; j < NN; j += 256) s += fabsf((float)attn[(size_t)i * NN + j]);
    red[threadIdx.x] = s;
    __syncthreads();
    for (int w = 128; w > 0; w >>= 1) {
        if (threadIdx.x < w) red[threadIdx.x] += red[threadIdx.x + w];
        __syncthreads();
    }
    if (threadIdx.x == 0) denom[i] = fmaxf(red[0], 1e-12f);
}

// hsT[n][j] = hs[j][n]   (128 x 4096 f16)
__global__ void transpose_f16(const _Float16* __restrict__ src, _Float16* __restrict__ dst) {
    const int j = blockIdx.x;
    const int n = threadIdx.x;  // 128 threads
    dst[(size_t)n * NN + j] = src[(size_t)j * DIM + n];
}

// xs[i][n] += (attn * hsT^T)[i][n] / denom[i]    (K = 4096)
__global__ void gemm_norm_axpy(const _Float16* __restrict__ attn, const _Float16* __restrict__ hsT,
                               const float* __restrict__ denom, float* __restrict__ xs) {
    const int i0 = blockIdx.x * 16;
    const int j0 = blockIdx.y * 16;
    const _Float16* Ab = attn + (size_t)i0 * NN;
    const _Float16* Bb = hsT + (size_t)j0 * NN;
    v8f acc = {};
    for (int k = 0; k < NN; k += 32) {
        __builtin_prefetch(Ab + k + 256, 0, 0);  // global_prefetch_b8
        v16h a = frag_load(Ab + k, NN);
        v16h b = frag_load(Bb + k, NN);
        acc = __builtin_amdgcn_wmma_f32_16x16x32_f16(false, a, false, b, (short)0, acc,
                                                     false, false);
    }
    int rb, col; frag_rc(rb, col);
#pragma unroll
    for (int r = 0; r < 8; ++r) {
        const int row = i0 + rb + r;
        xs[(size_t)row * DIM + j0 + col] += acc[r] / denom[row];
    }
}

// ---------------------------------------------------------------------------
// Embedding gather + conversions
// ---------------------------------------------------------------------------
__global__ void gather_embed(const int* __restrict__ idx, const float* __restrict__ emb,
                             float* __restrict__ of32, _Float16* __restrict__ of16) {
    const int i = blockIdx.x;
    const int n = threadIdx.x;  // 128
    const float v = emb[(size_t)idx[i] * DIM + n];
    if (of32) of32[(size_t)i * DIM + n] = v;
    if (of16) of16[(size_t)i * DIM + n] = (_Float16)v;
}

__global__ void f32_to_f16(const float* __restrict__ s, _Float16* __restrict__ d, int n) {
    const int i = blockIdx.x * 256 + threadIdx.x;
    if (i < n) d[i] = (_Float16)s[i];
}

__global__ void vec_add(const float* __restrict__ a, const float* __restrict__ b,
                        float* __restrict__ c, int n) {
    const int i = blockIdx.x * 256 + threadIdx.x;
    if (i < n) c[i] = a[i] + b[i];
}

// ---------------------------------------------------------------------------
// Sequential LSTM recurrence. Whh^T (512x128 f32 = 256KB) resident in the
// 320KB WGP LDS, transposed so each step's 512-lane read is bank-conflict
// free and h[k] is a bank broadcast. Pre-gates G = wv*Wih^T + b hoisted into
// a WMMA GEMM. Two blocks = forward / backward direction.
// ---------------------------------------------------------------------------
__global__ void lstm_dir(const float* __restrict__ G, const float* __restrict__ Whh,
                         float* __restrict__ hout) {
    extern __shared__ float sm[];
    float* WhhT = sm;                // [128 * 512]
    float* h_sh = sm + DIM * 512;    // [128]
    float* z_sh = h_sh + DIM;        // [512]
    const int d   = blockIdx.x;
    const int tid = threadIdx.x;     // 512 threads
    const float* W  = Whh + (size_t)d * 512 * DIM;
    const float* Gd = G + (size_t)d * MM * 512;
    float* hd = hout + (size_t)d * MM * DIM;

    for (int idx = tid; idx < 512 * DIM; idx += 512) {
        const int j = idx >> 7, k = idx & 127;
        WhhT[k * 512 + j] = W[idx];
    }
    if (tid < DIM) h_sh[tid] = 0.f;
    float c = 0.f;
    __syncthreads();

    for (int t = 0; t < MM; ++t) {
        const int tt = d ? (MM - 1 - t) : t;
        float acc = Gd[(size_t)tt * 512 + tid];
#pragma unroll 8
        for (int k = 0; k < DIM; ++k) acc += h_sh[k] * WhhT[k * 512 + tid];
        z_sh[tid] = acc;
        __syncthreads();
        if (tid < DIM) {
            const float zi = z_sh[tid], zf = z_sh[tid + 128];
            const float zg = z_sh[tid + 256], zo = z_sh[tid + 384];
            const float ig = 1.f / (1.f + __expf(-zi));
            const float fg = 1.f / (1.f + __expf(-zf));
            const float gg = tanhf(zg);
            const float og = 1.f / (1.f + __expf(-zo));
            c = fg * c + ig * gg;
            const float h = og * tanhf(c);
            h_sh[tid] = h;
            hd[(size_t)tt * DIM + tid] = h;
        }
        __syncthreads();
    }
}

// xs_rnn[t] = relu(concat(h_f[t], h_b[t])) as f16
__global__ void bilstm_concat_relu(const float* __restrict__ hbuf, _Float16* __restrict__ xs_rnn) {
    const int t = blockIdx.x;
    const int n = threadIdx.x;  // 256
    const float v = (n < DIM) ? hbuf[(size_t)t * DIM + n]
                              : hbuf[(size_t)(MM + t) * DIM + (n - DIM)];
    xs_rnn[(size_t)t * (2 * DIM) + n] = (_Float16)fmaxf(v, 0.f);
}

// ---------------------------------------------------------------------------
// Reductions over the weights matrix and final pooling / MLP
// ---------------------------------------------------------------------------
__global__ void row_mean(const float* __restrict__ W, float* __restrict__ out) {
    __shared__ float red[256];
    const int i = blockIdx.x;
    float s = 0.f;
    for (int j = threadIdx.x; j < MM; j += 256) s += W[(size_t)i * MM + j];
    red[threadIdx.x] = s;
    __syncthreads();
    for (int w = 128; w > 0; w >>= 1) {
        if (threadIdx.x < w) red[threadIdx.x] += red[threadIdx.x + w];
        __syncthreads();
    }
    if (threadIdx.x == 0) out[i] = red[0] / (float)MM;
}

__global__ void col_mean(const float* __restrict__ W, float* __restrict__ out) {
    const int j = blockIdx.x * 256 + threadIdx.x;
    float s = 0.f;
    for (int i = 0; i < NN; ++i) s += W[(size_t)i * MM + j];
    out[j] = s / (float)NN;
}

// cat[0:128)  = mean_i w_comp[i]*h[i,:]   ; cat[128:256) = mean_j w_prot[j]*hp[j,:]
__global__ void pool_vectors(const float* __restrict__ h, const float* __restrict__ hp,
                             const float* __restrict__ wc, const float* __restrict__ wp,
                             float* __restrict__ cat) {
    const int tid = threadIdx.x;  // 256
    const int n = tid & 127;
    const float* src = (tid < 128) ? h : hp;
    const float* wv  = (tid < 128) ? wc : wp;
    float s = 0.f;
    for (int i = 0; i < NN; ++i) s += wv[i] * src[(size_t)i * DIM + n];
    cat[tid] = s / (float)NN;
}

__global__ void final_mlp(const float* __restrict__ cat_in, const float* __restrict__ Wo,
                          const float* __restrict__ bo, const float* __restrict__ Wi,
                          const float* __restrict__ bi, float* __restrict__ out) {
    __shared__ float cur[256], nxt[256];
    const int t = threadIdx.x;  // 256
    cur[t] = cat_in[t];
    __syncthreads();
    for (int l = 0; l < 2; ++l) {
        const float* Wl = Wo + (size_t)l * 256 * 256;
        float s = bo[l * 256 + t];
        for (int k = 0; k < 256; ++k) s += cur[k] * Wl[(size_t)t * 256 + k];
        nxt[t] = fmaxf(s, 0.f);
        __syncthreads();
        cur[t] = nxt[t];
        __syncthreads();
    }
    if (t < 2) {
        float s = bi[t];
        for (int k = 0; k < 256; ++k) s += cur[k] * Wi[(size_t)t * 256 + k];
        out[t] = s;
    }
}

// ---------------------------------------------------------------------------
extern "C" void kernel_launch(void* const* d_in, const int* in_sizes, int n_in,
                              void* d_out, int out_size, void* d_ws, size_t ws_size,
                              hipStream_t stream) {
    (void)in_sizes; (void)n_in; (void)out_size; (void)ws_size;
    const int*   fingerprints = (const int*)d_in[0];
    const float* adjacency    = (const float*)d_in[1];
    const int*   words        = (const int*)d_in[2];
    const float* emb_fp       = (const float*)d_in[3];
    const float* emb_word     = (const float*)d_in[4];
    const float* Wg  = (const float*)d_in[5];
    const float* bg  = (const float*)d_in[6];
    const float* Wih = (const float*)d_in[7];
    const float* Whh = (const float*)d_in[8];
    const float* bih = (const float*)d_in[9];
    const float* bhh = (const float*)d_in[10];
    const float* Wc  = (const float*)d_in[11];
    const float* bc  = (const float*)d_in[12];
    const float* Wp  = (const float*)d_in[13];
    const float* bp  = (const float*)d_in[14];
    const float* Wo  = (const float*)d_in[15];
    const float* bo  = (const float*)d_in[16];
    const float* Wi  = (const float*)d_in[17];
    const float* bi  = (const float*)d_in[18];

    float* out         = (float*)d_out;
    float* weights_out = out + 2;  // [4096, 4096]

    // ---- workspace carve-out ----
    char* base = (char*)d_ws;
    size_t off = 0;
    auto alloc = [&](size_t bytes) -> void* {
        void* p = base + off;
        off = (off + bytes + 255) & ~(size_t)255;
        return p;
    };
    float*    xs       = (float*)   alloc((size_t)NN * DIM * 4);
    _Float16* xs_h     = (_Float16*)alloc((size_t)NN * DIM * 2);
    _Float16* hs_h     = (_Float16*)alloc((size_t)NN * DIM * 2);
    _Float16* hsT_h    = (_Float16*)alloc((size_t)DIM * NN * 2);
    _Float16* attn_h   = (_Float16*)alloc((size_t)NN * NN * 2);
    float*    denom    = (float*)   alloc((size_t)NN * 4);
    _Float16* Wg_h     = (_Float16*)alloc((size_t)3 * DIM * DIM * 2);
    _Float16* Wc_h     = (_Float16*)alloc((size_t)DIM * DIM * 2);
    _Float16* Wp_h     = (_Float16*)alloc((size_t)DIM * 2 * DIM * 2);
    _Float16* Wih_h    = (_Float16*)alloc((size_t)2 * 512 * DIM * 2);
    float*    bsum     = (float*)   alloc((size_t)2 * 512 * 4);
    _Float16* wv_h     = (_Float16*)alloc((size_t)MM * DIM * 2);
    float*    G        = (float*)   alloc((size_t)2 * MM * 512 * 4);
    float*    hbuf     = (float*)   alloc((size_t)2 * MM * DIM * 4);
    _Float16* xs_rnn_h = (_Float16*)alloc((size_t)MM * 2 * DIM * 2);
    float*    h_f32    = (float*)   alloc((size_t)NN * DIM * 4);
    _Float16* h_h      = (_Float16*)alloc((size_t)NN * DIM * 2);
    float*    hp_f32   = (float*)   alloc((size_t)MM * DIM * 4);
    _Float16* hp_h     = (_Float16*)alloc((size_t)MM * DIM * 2);
    float*    w_comp   = (float*)   alloc((size_t)NN * 4);
    float*    w_prot   = (float*)   alloc((size_t)MM * 4);
    float*    cat      = (float*)   alloc(256 * 4);

    // ---- one-time weight conversions ----
    f32_to_f16<<<(3 * DIM * DIM + 255) / 256, 256, 0, stream>>>(Wg, Wg_h, 3 * DIM * DIM);
    f32_to_f16<<<(DIM * DIM + 255) / 256, 256, 0, stream>>>(Wc, Wc_h, DIM * DIM);
    f32_to_f16<<<(DIM * 2 * DIM + 255) / 256, 256, 0, stream>>>(Wp, Wp_h, DIM * 2 * DIM);
    f32_to_f16<<<(2 * 512 * DIM + 255) / 256, 256, 0, stream>>>(Wih, Wih_h, 2 * 512 * DIM);
    vec_add<<<(2 * 512 + 255) / 256, 256, 0, stream>>>(bih, bhh, bsum, 2 * 512);

    // ---- embeddings ----
    gather_embed<<<NN, DIM, 0, stream>>>(fingerprints, emb_fp, xs, nullptr);
    gather_embed<<<MM, DIM, 0, stream>>>(words, emb_word, nullptr, wv_h);

    // ---- GAT layers ----
    for (int l = 0; l < 3; ++l) {
        f32_to_f16<<<(NN * DIM + 255) / 256, 256, 0, stream>>>(xs, xs_h, NN * DIM);
        gemm_bias_act<<<dim3(NN / 16, DIM / 16), 32, 0, stream>>>(
            xs_h, Wg_h + (size_t)l * DIM * DIM, bg + l * DIM,
            nullptr, hs_h, DIM, DIM, DIM, DIM, 1);
        transpose_f16<<<NN, DIM, 0, stream>>>(hs_h, hsT_h);
        gemm_attn<<<dim3(NN / 16, NN / 16), 32, 0, stream>>>(hs_h, adjacency, attn_h);
        row_abs_sum<<<NN, 256, 0, stream>>>(attn_h, denom);
        gemm_norm_axpy<<<dim3(NN / 16, DIM / 16), 32, 0, stream>>>(attn_h, hsT_h, denom, xs);
    }

    // ---- compound projection h = relu(x * Wc^T + bc) ----
    f32_to_f16<<<(NN * DIM + 255) / 256, 256, 0, stream>>>(xs, xs_h, NN * DIM);
    gemm_bias_act<<<dim3(NN / 16, DIM / 16), 32, 0, stream>>>(
        xs_h, Wc_h, bc, h_f32, h_h, DIM, DIM, DIM, DIM, 1);

    // ---- BiLSTM: pre-gates via WMMA, recurrence in LDS ----
    for (int d = 0; d < 2; ++d) {
        gemm_bias_act<<<dim3(MM / 16, 512 / 16), 32, 0, stream>>>(
            wv_h, Wih_h + (size_t)d * 512 * DIM, bsum + d * 512,
            G + (size_t)d * MM * 512, nullptr, DIM, DIM, DIM, 512, 0);
    }
    const size_t lstm_lds = (size_t)(DIM * 512 + DIM + 512) * sizeof(float);  // ~258.5 KB
    lstm_dir<<<2, 512, lstm_lds, stream>>>(G, Whh, hbuf);
    bilstm_concat_relu<<<MM, 256, 0, stream>>>(hbuf, xs_rnn_h);
    gemm_bias_act<<<dim3(MM / 16, DIM / 16), 32, 0, stream>>>(
        xs_rnn_h, Wp_h, bp, hp_f32, hp_h, 2 * DIM, 2 * DIM, 2 * DIM, DIM, 1);

    // ---- cross-attention weights = h * hp^T (model output #2) ----
    gemm_bias_act<<<dim3(NN / 16, MM / 16), 32, 0, stream>>>(
        h_h, hp_h, nullptr, weights_out, nullptr, DIM, DIM, DIM, MM, 0);

    // ---- pooling + output MLP ----
    row_mean<<<NN, 256, 0, stream>>>(weights_out, w_comp);
    col_mean<<<MM / 256, 256, 0, stream>>>(weights_out, w_prot);
    pool_vectors<<<1, 256, 0, stream>>>(h_f32, hp_f32, w_comp, w_prot, cat);
    final_mlp<<<1, 256, 0, stream>>>(cat, Wo, bo, Wi, bi, out);
}
